// CausalSelfAttention_26551487824025
// MI455X (gfx1250) — compile-verified, matrix operand order checked by reading, not run on hardware
//
#include <hip/hip_runtime.h>

// ---------------------------------------------------------------------------
// Types for CDNA5 WMMA (wave32): v_wmma_f32_16x16x32_bf16
// ---------------------------------------------------------------------------
typedef __attribute__((ext_vector_type(16))) __bf16        v16bf;
typedef __attribute__((ext_vector_type(8)))  __bf16        v8bf;
typedef __attribute__((ext_vector_type(8)))  float         v8f;
typedef __attribute__((ext_vector_type(4)))  float         v4f;
typedef __attribute__((ext_vector_type(4)))  unsigned int  v4u;
typedef __attribute__((ext_vector_type(4)))  unsigned int  u32x4;
typedef __attribute__((ext_vector_type(8)))  int           i32x8;
typedef __attribute__((ext_vector_type(4)))  int           i32x4;

#if __has_builtin(__builtin_amdgcn_tensor_load_to_lds)
#define USE_TDM 1
#else
#define USE_TDM 0
#endif

static __device__ __forceinline__ unsigned bfbits(float f) {
    union { float f; unsigned u; } x; x.f = f;
    return (x.u + 0x7FFFu + ((x.u >> 16) & 1u)) >> 16;     // RNE
}
static __device__ __forceinline__ __bf16 f2bf(float f) {
    unsigned short s = (unsigned short)bfbits(f);
    return __builtin_bit_cast(__bf16, s);
}
// pack two f32 -> packed bf16x2 (lo in [15:0], hi in [31:16])
static __device__ __forceinline__ unsigned pack_bf2(float lo, float hi) {
    return (bfbits(hi) << 16) | (bfbits(lo) & 0xFFFFu);
}
// pack two bf16 -> u32
static __device__ __forceinline__ unsigned pack2(__bf16 lo, __bf16 hi) {
    return ((unsigned)__builtin_bit_cast(unsigned short, hi) << 16) |
            (unsigned)__builtin_bit_cast(unsigned short, lo);
}

static __device__ __forceinline__ v8f wmma_bf16(v16bf a, v16bf b, v8f c) {
    return __builtin_amdgcn_wmma_f32_16x16x32_bf16(
        false, a, false, b, (short)0, c, false, false);
}

// Load a 16x32 bf16 A/B fragment from row-major [row][k] storage.
// ISA 7.12.2: lane<16 -> row=lane,   K = {0..7, 16..23}
//             lane>=16 -> row=lane-16, K = {8..15, 24..31}
static __device__ __forceinline__ v16bf load_frag(const __bf16* base, int stride, int lane) {
    const int m    = lane & 15;
    const int koff = (lane >> 4) << 3;
    const __bf16* p = base + (size_t)m * stride + koff;
    v8bf lo = *(const v8bf*)(p);
    v8bf hi = *(const v8bf*)(p + 16);
    v16bf r;
#pragma unroll
    for (int i = 0; i < 8; ++i) { r[i] = lo[i]; r[i + 8] = hi[i]; }
    return r;
}

// ---------------------------------------------------------------------------
// GEMM + bias:  Out[M,N] = A[M,K] * W[K,N] + bias.
// 128x64 block / 256 threads (8 waves); each wave computes a 32x32 macro-tile
// (2 A-frags x 2 B-frags -> 4 WMMAs per K-step).
// ---------------------------------------------------------------------------
template <bool A_F32, bool OUT_F32>
__global__ __launch_bounds__(256) void gemm_bias_kernel(
    const void* __restrict__ Aptr, const float* __restrict__ W,
    const float* __restrict__ bias, void* __restrict__ Out,
    int M, int N, int K)
{
    constexpr int BK = 32;
    __shared__ __align__(16) __bf16 As[128][BK + 8];  // [m][k], row = 80B
    __shared__ __align__(16) __bf16 Bs[64][BK + 8];   // [n][k] (W transposed)

    const int tid  = threadIdx.x;
    const int lane = tid & 31;
    const int wid  = tid >> 5;
    const int m0   = blockIdx.y * 128;
    const int n0   = blockIdx.x * 64;
    const int mg   = (wid & 3) * 32;     // wave m base within block (2 m-tiles)
    const int ng   = (wid >> 2) * 32;    // wave n base within block (2 n-tiles)

    v8f acc00 = {}, acc01 = {}, acc10 = {}, acc11 = {};

    for (int k0 = 0; k0 < K; k0 += BK) {
        // ---- stage A tile (128x32): 16 elems/thread, two b128 LDS stores ----
        {
            const int r  = tid >> 1;          // 0..127
            const int kc = (tid & 1) * 16;    // 0 or 16
            if constexpr (A_F32) {
                const float* src = (const float*)Aptr + (size_t)(m0 + r) * K + k0 + kc;
                v4f f0 = *(const v4f*)src;
                v4f f1 = *(const v4f*)(src + 4);
                v4f f2 = *(const v4f*)(src + 8);
                v4f f3 = *(const v4f*)(src + 12);
                v4u pk0, pk1;
                pk0[0] = pack_bf2(f0[0], f0[1]); pk0[1] = pack_bf2(f0[2], f0[3]);
                pk0[2] = pack_bf2(f1[0], f1[1]); pk0[3] = pack_bf2(f1[2], f1[3]);
                pk1[0] = pack_bf2(f2[0], f2[1]); pk1[1] = pack_bf2(f2[2], f2[3]);
                pk1[2] = pack_bf2(f3[0], f3[1]); pk1[3] = pack_bf2(f3[2], f3[3]);
                *(v4u*)&As[r][kc]     = pk0;
                *(v4u*)&As[r][kc + 8] = pk1;
                if (k0 + BK < K) __builtin_prefetch(src + BK, 0, 1);
            } else {
                const __bf16* src = (const __bf16*)Aptr + (size_t)(m0 + r) * K + k0 + kc;
                *(v4u*)&As[r][kc]     = *(const v4u*)src;
                *(v4u*)&As[r][kc + 8] = *(const v4u*)(src + 8);
                if (k0 + BK < K) __builtin_prefetch(src + BK, 0, 1);
            }
        }
        // ---- stage W transposed: Bs[n][k]; pack k-pairs into b32 stores ----
        {
            const int kp = tid >> 4;            // k rows 2kp, 2kp+1
            const int nc = (tid & 15) * 4;      // 4 n values
            const float* s0p = W + (size_t)(k0 + 2 * kp) * N + n0 + nc;
            const float* s1p = s0p + N;
            v4f w0 = *(const v4f*)s0p;
            v4f w1 = *(const v4f*)s1p;
#pragma unroll
            for (int i = 0; i < 4; ++i)
                *(unsigned*)&Bs[nc + i][2 * kp] = pack_bf2(w0[i], w1[i]);
            if (k0 + BK < K) __builtin_prefetch(s0p + (size_t)BK * N, 0, 1);
        }
        __syncthreads();

        v16bf fa0 = load_frag(&As[mg][0],      BK + 8, lane);
        v16bf fa1 = load_frag(&As[mg + 16][0], BK + 8, lane);
        v16bf fb0 = load_frag(&Bs[ng][0],      BK + 8, lane);
        v16bf fb1 = load_frag(&Bs[ng + 16][0], BK + 8, lane);
        acc00 = wmma_bf16(fa0, fb0, acc00);
        acc01 = wmma_bf16(fa0, fb1, acc01);
        acc10 = wmma_bf16(fa1, fb0, acc10);
        acc11 = wmma_bf16(fa1, fb1, acc11);
        __syncthreads();
    }

    const int hl = lane >> 4;
    const int nn = lane & 15;
#pragma unroll
    for (int r = 0; r < 8; ++r) {
        const int gm0 = m0 + mg + r + hl * 8;
        const int gm1 = gm0 + 16;
        const int gn0 = n0 + ng + nn;
        const int gn1 = gn0 + 16;
        const float v00 = acc00[r] + bias[gn0];
        const float v01 = acc01[r] + bias[gn1];
        const float v10 = acc10[r] + bias[gn0];
        const float v11 = acc11[r] + bias[gn1];
        if constexpr (OUT_F32) {
            float* O = (float*)Out;
            O[(size_t)gm0 * N + gn0] = v00;
            O[(size_t)gm0 * N + gn1] = v01;
            O[(size_t)gm1 * N + gn0] = v10;
            O[(size_t)gm1 * N + gn1] = v11;
        } else {
            __bf16* O = (__bf16*)Out;
            O[(size_t)gm0 * N + gn0] = f2bf(v00);
            O[(size_t)gm0 * N + gn1] = f2bf(v01);
            O[(size_t)gm1 * N + gn0] = f2bf(v10);
            O[(size_t)gm1 * N + gn1] = f2bf(v11);
        }
    }
}

// ---------------------------------------------------------------------------
// Flash-style causal attention (bf16 qkv -> bf16 y).
// Block = 64 q rows x 1 head x 1 batch, 128 threads = 4 waves.
// K tile staged by the Tensor Data Mover when available.
// ---------------------------------------------------------------------------
__global__ __launch_bounds__(128) void attn_kernel(
    const __bf16* __restrict__ qkv, __bf16* __restrict__ y)
{
    constexpr int S = 2048, D = 1024, TD = 3072, HD = 64;
    __shared__ __align__(16) __bf16 Ks[32][HD + 8];      // [key][feat]  row=144B
    __shared__ __align__(16) __bf16 Vs[HD][32 + 8];      // [feat][key]  row=80B
    __shared__ __align__(16) __bf16 Ps[4][16][32 + 8];   // per-wave P tile

    const int tid  = threadIdx.x;
    const int lane = tid & 31;
    const int wid  = tid >> 5;
    const int qb   = blockIdx.x;
    const int h    = blockIdx.y;
    const int b    = blockIdx.z;
    const int q0   = qb * 64 + wid * 16;
    const size_t rowbase = (size_t)(b * S) * TD;

    const __bf16* qbase = qkv + rowbase + (size_t)q0 * TD + h * HD;
    const v16bf qa0 = load_frag(qbase,      TD, lane);
    const v16bf qa1 = load_frag(qbase + 32, TD, lane);

    v8f o0 = {}, o1 = {}, o2 = {}, o3 = {};
    float mi[8], li[8];
#pragma unroll
    for (int r = 0; r < 8; ++r) { mi[r] = -3e38f; li[r] = 0.0f; }

    const int hl = lane >> 4;
    const int nn = lane & 15;
    const int kb_end = qb * 2 + 1;

#if USE_TDM
    // D# group0/group1 (ISA 08_async_tensor §8.3/8.4), invariant parts.
    // 2D tile: tile_dim0=64 bf16 (128B), tile_dim1=32 rows, row stride 3072,
    // LDS pad: every 32 DWORDs (code 4) insert 4 DWORDs (code 3) -> 144B rows.
    u32x4 g0b = {};
    g0b[0] = 1u;                                        // count=1, user mode
    g0b[1] = (unsigned)(size_t)(&Ks[0][0]);             // lds_addr (low 32 = LDS offset)
    i32x8 g1 = {};
    g1[0] = (1 << 16) | (1 << 20) | (4 << 22) | (3 << 25);  // data_size=2B, pad_en, interval, amount
    g1[1] = (64 << 16);                                 // tensor_dim0 = 64
    g1[2] = (32 << 16);                                 // tensor_dim1 = 32
    g1[3] = (64 << 16);                                 // tile_dim0 = 64
    g1[4] = 32;                                         // tile_dim1 = 32, tile_dim2 = 0
    g1[5] = TD;                                         // tensor_dim0_stride = 3072
    g1[6] = 0; g1[7] = 0;
    const i32x4 gz4 = {0, 0, 0, 0};
    const i32x8 gz8 = {0, 0, 0, 0, 0, 0, 0, 0};
#endif

    for (int kb = 0; kb <= kb_end; ++kb) {
        // ---- stage K tile ----
#if USE_TDM
        if (wid == 0) {
            const __bf16* ksrc0 = qkv + rowbase + (size_t)(kb * 32) * TD + D + h * HD;
            unsigned long long ga = (unsigned long long)(size_t)ksrc0;
            u32x4 g0 = g0b;
            g0[2] = (unsigned)(ga & 0xFFFFFFFFull);
            g0[3] = (unsigned)((ga >> 32) & 0x01FFFFFFull) | (2u << 30);  // type=2
            __builtin_amdgcn_tensor_load_to_lds(g0, g1, gz4, gz4, gz8, 0);
            __builtin_amdgcn_s_wait_tensorcnt(0);
        }
#else
        {
            const int kr = tid >> 2;
            const int f2 = (tid & 3) * 16;
            const __bf16* ksrc = qkv + rowbase + (size_t)(kb * 32 + kr) * TD + D + h * HD + f2;
            *(v4u*)&Ks[kr][f2]     = *(const v4u*)ksrc;
            *(v4u*)&Ks[kr][f2 + 8] = *(const v4u*)(ksrc + 8);
        }
#endif
        // ---- stage V transposed [feat][key]; pack key-pairs into b32 ----
        {
            const int kp = tid >> 3;            // keys 2kp, 2kp+1
            const int fc = (tid & 7) * 8;       // 8 feats
            const __bf16* v0p = qkv + rowbase + (size_t)(kb * 32 + 2 * kp) * TD + 2 * D + h * HD + fc;
            const __bf16* v1p = v0p + TD;
            v8bf a0 = *(const v8bf*)v0p;
            v8bf a1 = *(const v8bf*)v1p;
#pragma unroll
            for (int i = 0; i < 8; ++i)
                *(unsigned*)&Vs[fc + i][2 * kp] = pack2(a0[i], a1[i]);
        }
        __syncthreads();

        // ---- S = Q * K^T ----
        v8f s0 = {}, s1 = {};
        {
            v16bf b00 = load_frag(&Ks[0][0],   HD + 8, lane);
            v16bf b01 = load_frag(&Ks[0][32],  HD + 8, lane);
            v16bf b10 = load_frag(&Ks[16][0],  HD + 8, lane);
            v16bf b11 = load_frag(&Ks[16][32], HD + 8, lane);
            s0 = wmma_bf16(qa0, b00, s0);
            s0 = wmma_bf16(qa1, b01, s0);
            s1 = wmma_bf16(qa0, b10, s1);
            s1 = wmma_bf16(qa1, b11, s1);
        }

        // ---- scale, causal mask, online softmax ----
        float scale_r[8];
#pragma unroll
        for (int r = 0; r < 8; ++r) {
            const int qrow = q0 + r + hl * 8;
            const int key0 = kb * 32 + nn;
            const int key1 = key0 + 16;
            float a0 = s0[r] * 0.125f;
            float a1 = s1[r] * 0.125f;
            if (key0 > qrow) a0 = -3e38f;
            if (key1 > qrow) a1 = -3e38f;

            float mx = fmaxf(a0, a1);
#pragma unroll
            for (int msk = 8; msk >= 1; msk >>= 1)
                mx = fmaxf(mx, __shfl_xor(mx, msk, 32));
            const float mnew = fmaxf(mi[r], mx);
            const float sc = __expf(mi[r] - mnew);
            const float p0 = __expf(a0 - mnew);
            const float p1 = __expf(a1 - mnew);
            float rs = p0 + p1;
#pragma unroll
            for (int msk = 8; msk >= 1; msk >>= 1)
                rs += __shfl_xor(rs, msk, 32);
            li[r] = li[r] * sc + rs;
            mi[r] = mnew;
            scale_r[r] = sc;

            const int m = r + hl * 8;
            Ps[wid][m][nn]      = f2bf(p0);
            Ps[wid][m][16 + nn] = f2bf(p1);
        }
#pragma unroll
        for (int r = 0; r < 8; ++r) {
            o0[r] *= scale_r[r]; o1[r] *= scale_r[r];
            o2[r] *= scale_r[r]; o3[r] *= scale_r[r];
        }

        // ---- O += P * V ----
        v16bf pa = load_frag(&Ps[wid][0][0], 32 + 8, lane);
        o0 = wmma_bf16(pa, load_frag(&Vs[0][0],  32 + 8, lane), o0);
        o1 = wmma_bf16(pa, load_frag(&Vs[16][0], 32 + 8, lane), o1);
        o2 = wmma_bf16(pa, load_frag(&Vs[32][0], 32 + 8, lane), o2);
        o3 = wmma_bf16(pa, load_frag(&Vs[48][0], 32 + 8, lane), o3);
        __syncthreads();
    }

#pragma unroll
    for (int r = 0; r < 8; ++r) {
        const float inv = 1.0f / li[r];
        const int qrow = q0 + r + hl * 8;
        const size_t orow = ((size_t)(b * S) + qrow) * D + h * HD;
        y[orow +  0 + nn] = f2bf(o0[r] * inv);
        y[orow + 16 + nn] = f2bf(o1[r] * inv);
        y[orow + 32 + nn] = f2bf(o2[r] * inv);
        y[orow + 48 + nn] = f2bf(o3[r] * inv);
    }
}

// ---------------------------------------------------------------------------
// Launcher
// ---------------------------------------------------------------------------
extern "C" void kernel_launch(void* const* d_in, const int* in_sizes, int n_in,
                              void* d_out, int out_size, void* d_ws, size_t ws_size,
                              hipStream_t stream) {
    (void)in_sizes; (void)n_in; (void)out_size; (void)ws_size;
    const float* x      = (const float*)d_in[0];
    const float* w_attn = (const float*)d_in[1];
    const float* b_attn = (const float*)d_in[2];
    const float* w_proj = (const float*)d_in[3];
    const float* b_proj = (const float*)d_in[4];
    float* out = (float*)d_out;

    constexpr int B = 4, S = 2048, D = 1024;
    const int M = B * S;

    __bf16* qkv = reinterpret_cast<__bf16*>(d_ws);   // [M, 3D] bf16 (48 MB)
    __bf16* yb  = qkv + (size_t)M * 3 * D;           // [M, D]  bf16 (16 MB)

    dim3 g1(3 * D / 64, M / 128);
    gemm_bias_kernel<true, false><<<g1, 256, 0, stream>>>(x, w_attn, b_attn, qkv, M, 3 * D, D);

    dim3 g2(S / 64, 16, B);
    attn_kernel<<<g2, 128, 0, stream>>>(qkv, yb);

    dim3 g3(D / 64, M / 128);
    gemm_bias_kernel<false, true><<<g3, 256, 0, stream>>>(yb, w_proj, b_proj, out, M, D, D);
}
